// CustomConv_54589034332908
// MI455X (gfx1250) — matrix operand order repaired and checked by reading, not compiled, not run on hardware
//
#include <hip/hip_runtime.h>
#include <stdint.h>

// ---- vector typedefs (CDNA5 WMMA operand shapes, wave32) ----
typedef _Float16 v16h __attribute__((ext_vector_type(16)));
typedef _Float16 h8   __attribute__((ext_vector_type(8)));
typedef float    v8f  __attribute__((ext_vector_type(8)));
// ---- TDM descriptor groups (6-arg builtin on this toolchain) ----
typedef unsigned int u32x4 __attribute__((ext_vector_type(4)));
typedef int          i32x8 __attribute__((ext_vector_type(8)));
typedef int          i32x4 __attribute__((ext_vector_type(4)));

#define BATCH 32
#define IC    128
#define OC    256
#define WDIM  56
#define HW    3136        // 56*56
#define KT    32          // ic-chunk per K-step (matches 16x16x32 WMMA)
#define MT    128         // M tile (output channels per block)
#define NT    64          // N tile (3136 = 49*64, never crosses an image)
#define LDA   40          // LDS row stride (halfs): 32 data + 8 pad = 80B rows (TDM pad reproduces this)
#define LDB   40
#define NSTEP 36          // 9 taps * 4 ic-chunks

// ---------- prepass: w[oc][ic][3][3] f32  ->  wt[tap][oc][ic] f16 ----------
__global__ void prep_weights(const float* __restrict__ w, _Float16* __restrict__ wt)
{
    int idx = blockIdx.x * 256 + threadIdx.x;        // 0 .. 9*256*128-1 (exact)
    int ic  = idx & (IC - 1);
    int oc  = (idx >> 7) & (OC - 1);
    int tap = idx >> 15;
    wt[((size_t)tap * OC + oc) * IC + ic] = (_Float16)w[((size_t)oc * IC + ic) * 9 + tap];
}

// ---------- main conv kernel: implicit GEMM, tap-major K loop, TDM A-staging ----------
__launch_bounds__(256)
__global__ void conv3x3_igemm_wmma_f16(const float* __restrict__ x,
                                       const _Float16* __restrict__ wt,
                                       const float* __restrict__ bias,
                                       float* __restrict__ out)
{
    __shared__ __align__(16) _Float16 As[2][MT * LDA];   // 2 x 10240 B (TDM destination)
    __shared__ __align__(16) _Float16 Bs[2][NT * LDB];   // 2 x  5120 B

    const int tid   = threadIdx.x;
    const int lane  = tid & 31;
    const int wave  = tid >> 5;
    const int l16   = lane & 15;
    const int halfK = lane >> 4;

    const int gM   = blockIdx.y * MT;
    const int bImg = blockIdx.x / 49;
    const int rem0 = (blockIdx.x % 49) * NT;

    const int waveM = (wave >> 1) * 32;   // 4 x 2 wave grid over 128x64 tile
    const int waveN = (wave & 1) * 32;

    // ---- per-thread staging coordinates (hoisted once) ----
    const int nB   = tid & 63;            // pixel within tile
    const int icg  = tid >> 6;            // ic subgroup 0..3 (8 ic each)
    const int remB = rem0 + nB;
    const int ohB  = remB / WDIM;
    const int owB  = remB - ohB * WDIM;
    const float* xbase0 = x + (size_t)bImg * IC * HW;

    // ---- constant part of the TDM descriptor group 1 (ISA D# layout) ----
    // data_size=1(2B) | pad_enable | pad_interval=3 (16 DWORDs = 64B) | pad_amount=3 (4 DWORDs = 16B)
    const i32x8 g1 = { (int)((1u << 16) | (1u << 20) | (3u << 22) | (3u << 25)),
                       (int)((unsigned)IC << 16),     // tensor_dim0 = 128 (bits 79:48)
                       (int)((unsigned)OC << 16),     // tensor_dim1 = 256 (bits 111:80)
                       (int)((unsigned)KT << 16),     // tile_dim0  = 32  (bits 127:112)
                       (int)MT,                       // tile_dim1  = 128 (bits 143:128)
                       (int)IC,                       // tensor_dim0_stride = 128 (bits 207:160)
                       0, 0 };
    const i32x4 gz  = { 0, 0, 0, 0 };                 // groups 2/3 unused (2-D tensor)
    const i32x8 gz8 = { 0, 0, 0, 0, 0, 0, 0, 0 };     // extra group slot (unused, count=1)

    v8f acc00 = {}; v8f acc01 = {}; v8f acc10 = {}; v8f acc11 = {};

    // ---- staging for step s into buffer p ----
    auto stage = [&](int s, int p) {
        const int tap = s >> 2;
        const int icc = (s & 3) * KT;
        // --- A: one TDM descriptor moves the whole 128x32 f16 tile (wave 0 only) ---
        if (wave == 0) {
            uint64_t ga  = (uint64_t)(size_t)(wt + ((size_t)tap * OC + gM) * IC + icc);
            uint32_t lda = (uint32_t)(size_t)&As[p][0];
            u32x4 g0 = { 1u,                                   // count=1 (valid descriptor)
                         lda,                                  // lds_addr (bytes)
                         (uint32_t)ga,                         // global_addr[31:0]
                         (uint32_t)(ga >> 32) | (2u << 30) };  // global_addr[56:32] | type=2
            __builtin_amdgcn_tensor_load_to_lds(g0, g1, gz, gz, gz8, 0);
        }
        // --- B: im2col gather for this tap, 8 strided f32 loads -> h8 store ---
        const int dh = tap / 3 - 1;
        const int dw = tap - (tap / 3) * 3 - 1;
        const int ih = ohB + dh;
        const int iw = owB + dw;
        const bool ok = ((unsigned)ih < (unsigned)WDIM) & ((unsigned)iw < (unsigned)WDIM);
        const float* xb = xbase0 + (size_t)(icc + icg * 8) * HW + ih * WDIM + iw;
        float v[8];
        #pragma unroll
        for (int j = 0; j < 8; ++j)
            v[j] = ok ? xb[(size_t)j * HW] : 0.0f;
        h8 hv = { (_Float16)v[0], (_Float16)v[1], (_Float16)v[2], (_Float16)v[3],
                  (_Float16)v[4], (_Float16)v[5], (_Float16)v[6], (_Float16)v[7] };
        *(h8*)&Bs[p][nB * LDB + icg * 8] = hv;
    };

    // ---- compute 4 WMMAs from buffer p ----
    auto compute = [&](int p) {
        const _Float16* a0p = &As[p][(waveM +  0 + l16) * LDA];
        const _Float16* a1p = &As[p][(waveM + 16 + l16) * LDA];
        h8 a0lo = *(const h8*)(a0p + 8 * halfK);
        h8 a0hi = *(const h8*)(a0p + 16 + 8 * halfK);
        h8 a1lo = *(const h8*)(a1p + 8 * halfK);
        h8 a1hi = *(const h8*)(a1p + 16 + 8 * halfK);
        v16h a0 = __builtin_shufflevector(a0lo, a0hi, 0,1,2,3,4,5,6,7,8,9,10,11,12,13,14,15);
        v16h a1 = __builtin_shufflevector(a1lo, a1hi, 0,1,2,3,4,5,6,7,8,9,10,11,12,13,14,15);
        const _Float16* b0p = &Bs[p][(waveN +  0 + l16) * LDB + 8 * halfK];
        const _Float16* b1p = &Bs[p][(waveN + 16 + l16) * LDB + 8 * halfK];
        h8 b0lo = *(const h8*)(b0p);
        h8 b0hi = *(const h8*)(b0p + 16);
        h8 b1lo = *(const h8*)(b1p);
        h8 b1hi = *(const h8*)(b1p + 16);
        v16h b0 = __builtin_shufflevector(b0lo, b0hi, 0,1,2,3,4,5,6,7,8,9,10,11,12,13,14,15);
        v16h b1 = __builtin_shufflevector(b1lo, b1hi, 0,1,2,3,4,5,6,7,8,9,10,11,12,13,14,15);

        acc00 = __builtin_amdgcn_wmma_f32_16x16x32_f16(false, a0, false, b0, (short)0, acc00, false, false);
        acc01 = __builtin_amdgcn_wmma_f32_16x16x32_f16(false, a0, false, b1, (short)0, acc01, false, false);
        acc10 = __builtin_amdgcn_wmma_f32_16x16x32_f16(false, a1, false, b0, (short)0, acc10, false, false);
        acc11 = __builtin_amdgcn_wmma_f32_16x16x32_f16(false, a1, false, b1, (short)0, acc11, false, false);
    };

    // ---- double-buffered pipeline (TDM for step s+1 overlaps WMMAs of step s) ----
    stage(0, 0);
    for (int s = 0; s < NSTEP; ++s) {
        __builtin_amdgcn_s_wait_tensorcnt((short)0);   // A tile for step s landed in LDS
        __syncthreads();                               // ... and visible to all waves
        if (s + 1 < NSTEP) stage(s + 1, (s + 1) & 1);
        compute(s & 1);
        __syncthreads();                               // buf (s&1) free for restage at s+2
    }

    // ---- store: D layout VGPR r -> M = r + 8*halfK, N = l16; fuse bias ----
    #define STORE_TILE(ACC, MI, NI)                                                   \
        {                                                                             \
            _Pragma("unroll")                                                         \
            for (int r = 0; r < 8; ++r) {                                             \
                int oc  = gM + waveM + (MI)*16 + r + 8 * halfK;                       \
                int rem = rem0 + waveN + (NI)*16 + l16;                               \
                out[((size_t)bImg * OC + oc) * HW + rem] = ACC[r] + bias[oc];         \
            }                                                                         \
        }
    STORE_TILE(acc00, 0, 0)
    STORE_TILE(acc01, 0, 1)
    STORE_TILE(acc10, 1, 0)
    STORE_TILE(acc11, 1, 1)
    #undef STORE_TILE
}

extern "C" void kernel_launch(void* const* d_in, const int* in_sizes, int n_in,
                              void* d_out, int out_size, void* d_ws, size_t ws_size,
                              hipStream_t stream) {
    const float* x    = (const float*)d_in[0];   // (32,128,56,56) f32
    const float* wgt  = (const float*)d_in[1];   // (256,128,3,3)  f32
    const float* bias = (const float*)d_in[2];   // (256,)         f32
    float* out = (float*)d_out;                  // (32,256,56,56) f32

    _Float16* wt = (_Float16*)d_ws;              // 9*256*128 f16 = 576 KB scratch

    prep_weights<<<dim3(9 * OC * IC / 256), dim3(256), 0, stream>>>(wgt, wt);

    dim3 grid(49 * BATCH, OC / MT, 1);           // 1568 x 2 blocks
    dim3 block(256, 1, 1);                       // 8 waves (wave32)
    conv3x3_igemm_wmma_f16<<<grid, block, 0, stream>>>(x, wt, bias, out);
}
